// MOE_layers_attention_42013370090168
// MI455X (gfx1250) — compile-verified, matrix operand order checked by reading, not run on hardware
//
#include <hip/hip_runtime.h>
#include <hip/hip_bf16.h>
#include <math.h>

#define NHEADS 8
#define HD 64
#define BSZ 8
#define CH 64
#define NSEG 64
#define DMODEL 512
#define DFFN 2048
#define FACTOR 10
#define NLAYERS 2
#define GROUPS (BSZ * NSEG)          // 512
#define ROWS (GROUPS * CH)           // 32768
#define BUFROWS (GROUPS * FACTOR)    // 5120
#define QRROWS (NSEG * FACTOR)       // 640
#define BN_EPS 1e-5f
#define MLP_CHUNK 8192

typedef __attribute__((ext_vector_type(16))) __bf16 v16bf;
typedef __attribute__((ext_vector_type(8)))  float v8f;
typedef __attribute__((ext_vector_type(8)))  unsigned short v8us;

union FragU { v16bf v; v8us h[2]; };

__device__ __forceinline__ float bf2f(unsigned short u) {
  unsigned int x = ((unsigned int)u) << 16;
  return __builtin_bit_cast(float, x);
}
__device__ __forceinline__ unsigned short f2bf(float f) {
  unsigned int u = __builtin_bit_cast(unsigned int, f);
  u += 0x7FFFu + ((u >> 16) & 1u);            // round-to-nearest-even
  return (unsigned short)(u >> 16);
}

// One 16B async global->LDS copy per lane (ASYNCcnt path, bypasses VGPRs).
// IOFFSET is added to BOTH the LDS and global addresses (ISA 08 §4.4), which
// matches our tiling: LDS row slice and global row slice advance together.
#define ASYNC_CP16(LDSOFF, GADDR, OFF)                                        \
  asm volatile("global_load_async_to_lds_b128 %0, %1, off offset:" #OFF       \
               :: "v"(LDSOFF), "v"(GADDR) : "memory")

// ---------------------------------------------------------------------------
// bf16 WMMA GEMM: C[M,N] = A[M,K] * Bt[N,K]^T + bias, optional GELU,
// optional f32 and/or bf16 outputs. M,N multiples of 128; K multiple of 32.
// 128 threads = 4 waves (2x2), per-wave 64x64 tile, v_wmma_f32_16x16x32_bf16.
// Double-buffered LDS, async copies overlap the 16-WMMA compute block,
// one barrier + one s_wait_asynccnt per K-step.
// ---------------------------------------------------------------------------
#define LDSW 40   // 32 + 8 pad (80B row stride = 20 dwords -> conflict-free b128)

__global__ __launch_bounds__(128) void k_gemm_bf16(
    const unsigned short* __restrict__ A,
    const unsigned short* __restrict__ Bt,
    const float* __restrict__ bias,
    float* __restrict__ Cf,
    unsigned short* __restrict__ Cb,
    int M, int N, int K, int dogelu)
{
  __shared__ alignas(16) unsigned short As[2][128 * LDSW];
  __shared__ alignas(16) unsigned short Bs[2][128 * LDSW];

  const int tid  = threadIdx.x;
  const int lane = tid & 31;
  const int wave = tid >> 5;
  const int wm = (wave >> 1) * 64;      // wave row offset in block tile
  const int wn = (wave & 1) * 64;       // wave col offset
  const int ml = lane & 15;             // row/col within 16x16 frag
  const int kh = lane >> 4;             // K-half select (ISA 16-bit A layout)
  const int kb = kh * 8;

  const long rowBase = (long)blockIdx.y * 128;
  const long colBase = (long)blockIdx.x * 128;

  v8f acc[4][4] = {};

  const unsigned short* gA = A  + (rowBase + tid) * (long)K;
  const unsigned short* gB = Bt + (colBase + tid) * (long)K;

  // flat LDS pointers: low 32 bits == LDS byte address (ISA 10.2 aperture map)
  const unsigned laBase0 = (unsigned)(unsigned long long)&As[0][tid * LDSW];
  const unsigned laBase1 = (unsigned)(unsigned long long)&As[1][tid * LDSW];
  const unsigned lbBase0 = (unsigned)(unsigned long long)&Bs[0][tid * LDSW];
  const unsigned lbBase1 = (unsigned)(unsigned long long)&Bs[1][tid * LDSW];

  auto stage = [&](int buf, int k0) {
    const unsigned la = buf ? laBase1 : laBase0;
    const unsigned lb = buf ? lbBase1 : lbBase0;
    const unsigned long long ga = (unsigned long long)(gA + k0);
    const unsigned long long gb = (unsigned long long)(gB + k0);
    ASYNC_CP16(la, ga, 0);  ASYNC_CP16(la, ga, 16);
    ASYNC_CP16(la, ga, 32); ASYNC_CP16(la, ga, 48);
    ASYNC_CP16(lb, gb, 0);  ASYNC_CP16(lb, gb, 16);
    ASYNC_CP16(lb, gb, 32); ASYNC_CP16(lb, gb, 48);
  };

  stage(0, 0);
  asm volatile("s_wait_asynccnt 0x0" ::: "memory");
  __syncthreads();

  int cur = 0;
  for (int k0 = 0; k0 < K; k0 += 32) {
    if (k0 + 32 < K) stage(cur ^ 1, k0 + 32);   // prefetch next tile (async)

    const unsigned short* Ac = As[cur];
    const unsigned short* Bc = Bs[cur];
    FragU fa[4], fb[4];
#pragma unroll
    for (int i = 0; i < 4; ++i) {
      const unsigned short* p = &Ac[(wm + i * 16 + ml) * LDSW];
      fa[i].h[0] = *(const v8us*)(p + kb);          // K = kb..kb+7
      fa[i].h[1] = *(const v8us*)(p + 16 + kb);     // K = 16+kb..
    }
#pragma unroll
    for (int j = 0; j < 4; ++j) {
      const unsigned short* p = &Bc[(wn + j * 16 + ml) * LDSW];
      fb[j].h[0] = *(const v8us*)(p + kb);
      fb[j].h[1] = *(const v8us*)(p + 16 + kb);
    }
#pragma unroll
    for (int i = 0; i < 4; ++i)
#pragma unroll
      for (int j = 0; j < 4; ++j)
        acc[i][j] = __builtin_amdgcn_wmma_f32_16x16x32_bf16(
            false, fa[i].v, false, fb[j].v, (short)0, acc[i][j], false, false);

    asm volatile("s_wait_asynccnt 0x0" ::: "memory");
    __syncthreads();
    cur ^= 1;
  }

  // epilogue: C/D layout — lane%16 = col, vgpr e -> row = e + (lane/16)*8
#pragma unroll
  for (int j = 0; j < 4; ++j) {
    const long col = colBase + wn + j * 16 + ml;
    const float bv = bias ? bias[col] : 0.0f;
#pragma unroll
    for (int i = 0; i < 4; ++i) {
#pragma unroll
      for (int e = 0; e < 8; ++e) {
        const long row = rowBase + wm + i * 16 + kh * 8 + e;
        float x = acc[i][j][e] + bv;
        if (dogelu) x = 0.5f * x * (1.0f + erff(x * 0.70710678118654752f));
        const long idx = row * (long)N + col;
        if (Cf) Cf[idx] = x;
        if (Cb) Cb[idx] = f2bf(x);
      }
    }
  }
}

// ---------------------------------------------------------------------------
// layout / conversion kernels
// ---------------------------------------------------------------------------
__global__ void k_rearrange_in(const float* __restrict__ x,  // [B,C,N,D]
                               float* __restrict__ dsF,      // rows (b,n,c)
                               unsigned short* __restrict__ dsB)
{
  const long total = (long)ROWS * DMODEL;
  for (long i = (long)blockIdx.x * blockDim.x + threadIdx.x; i < total;
       i += (long)gridDim.x * blockDim.x) {
    long d = i & (DMODEL - 1);
    long r = i >> 9;               // /DMODEL
    long c = r % CH; r /= CH;
    long n = r % NSEG;
    long b = r / NSEG;
    float v = x[(((b * CH + c) * NSEG + n) << 9) + d];
    dsF[i] = v;
    dsB[i] = f2bf(v);
  }
}

__global__ void k_rearrange_out(const float* __restrict__ e,  // rows (b,n,c)
                                float* __restrict__ out)      // [B,C,N,D]
{
  const long total = (long)ROWS * DMODEL;
  for (long i = (long)blockIdx.x * blockDim.x + threadIdx.x; i < total;
       i += (long)gridDim.x * blockDim.x) {
    long d = i & (DMODEL - 1);
    long r = i >> 9;
    long n = r % NSEG; r /= NSEG;
    long c = r % CH;
    long b = r / CH;
    out[i] = e[(((b * NSEG + n) * CH + c) << 9) + d];
  }
}

__global__ void k_wtrans(const float* __restrict__ W,         // [K,N]
                         unsigned short* __restrict__ Wt,     // [N,K] bf16
                         int K, int N, long total)
{
  for (long i = (long)blockIdx.x * blockDim.x + threadIdx.x; i < total;
       i += (long)gridDim.x * blockDim.x) {
    long k = i / N, n = i % N;
    Wt[n * (long)K + k] = f2bf(W[i]);
  }
}

__global__ void k_cvt_bf16(const float* __restrict__ s,
                           unsigned short* __restrict__ d, long total)
{
  for (long i = (long)blockIdx.x * blockDim.x + threadIdx.x; i < total;
       i += (long)gridDim.x * blockDim.x)
    d[i] = f2bf(s[i]);
}

// ---------------------------------------------------------------------------
// router attention (buf): per (group g=b*NSEG+n, head): Q[10,64] x K[64,64]
// softmax over 64 -> out[10,64]. One 64-thread block per (g,h).
// ---------------------------------------------------------------------------
__global__ __launch_bounds__(64) void k_attn_router(
    const unsigned short* __restrict__ Qr,   // [QRROWS, D] rows (n,f)
    const unsigned short* __restrict__ Kd,   // [ROWS, D]  rows (g,c)
    const unsigned short* __restrict__ Vd,
    unsigned short* __restrict__ Obuf)       // [BUFROWS, D] rows (g,f)
{
  const int g = blockIdx.x, h = blockIdx.y;
  const int n = g % NSEG;
  const int t = threadIdx.x;                 // t = channel index c / out dim d
  __shared__ float sP[FACTOR][CH];
  __shared__ float sV[CH][HD];

  float kr[HD];
  const unsigned short* kp = Kd + ((long)(g * CH + t)) * DMODEL + h * HD;
  const unsigned short* vp = Vd + ((long)(g * CH + t)) * DMODEL + h * HD;
#pragma unroll
  for (int e = 0; e < HD; ++e) { kr[e] = bf2f(kp[e]); sV[t][e] = bf2f(vp[e]); }

  for (int f = 0; f < FACTOR; ++f) {
    const unsigned short* qp = Qr + ((long)(n * FACTOR + f)) * DMODEL + h * HD;
    float s = 0.0f;
#pragma unroll
    for (int e = 0; e < HD; ++e) s += bf2f(qp[e]) * kr[e];
    sP[f][t] = s * 0.125f;                   // 1/sqrt(64)
  }
  __syncthreads();
  if (t < FACTOR) {
    float mx = -1e30f;
    for (int c = 0; c < CH; ++c) mx = fmaxf(mx, sP[t][c]);
    float sum = 0.0f;
    for (int c = 0; c < CH; ++c) { float p = __expf(sP[t][c] - mx); sP[t][c] = p; sum += p; }
    float inv = 1.0f / sum;
    for (int c = 0; c < CH; ++c) sP[t][c] *= inv;
  }
  __syncthreads();
  for (int f = 0; f < FACTOR; ++f) {
    float o = 0.0f;
#pragma unroll
    for (int c = 0; c < CH; ++c) o += sP[f][c] * sV[c][t];
    Obuf[((long)(g * FACTOR + f)) * DMODEL + h * HD + t] = f2bf(o);
  }
}

// ---------------------------------------------------------------------------
// recv attention: per (g,h): Q[64,64] x K[10,64], softmax over 10 -> out[64,64]
// ---------------------------------------------------------------------------
__global__ __launch_bounds__(64) void k_attn_recv(
    const unsigned short* __restrict__ Qd,   // [ROWS, D] rows (g,c)
    const unsigned short* __restrict__ Kb,   // [BUFROWS, D] rows (g,f)
    const unsigned short* __restrict__ Vb,
    unsigned short* __restrict__ Orecv)      // [ROWS, D]
{
  const int g = blockIdx.x, h = blockIdx.y;
  const int t = threadIdx.x;                 // query row c
  __shared__ float sK[FACTOR][HD];
  __shared__ float sV[FACTOR][HD];
  for (int idx = t; idx < FACTOR * HD; idx += 64) {
    int f = idx >> 6, e = idx & 63;
    sK[f][e] = bf2f(Kb[((long)(g * FACTOR + f)) * DMODEL + h * HD + e]);
    sV[f][e] = bf2f(Vb[((long)(g * FACTOR + f)) * DMODEL + h * HD + e]);
  }
  float q[HD];
  const unsigned short* qp = Qd + ((long)(g * CH + t)) * DMODEL + h * HD;
#pragma unroll
  for (int e = 0; e < HD; ++e) q[e] = bf2f(qp[e]);
  __syncthreads();

  float p[FACTOR], mx = -1e30f;
  for (int f = 0; f < FACTOR; ++f) {
    float s = 0.0f;
#pragma unroll
    for (int e = 0; e < HD; ++e) s += q[e] * sK[f][e];
    p[f] = s * 0.125f;
    mx = fmaxf(mx, p[f]);
  }
  float sum = 0.0f;
  for (int f = 0; f < FACTOR; ++f) { p[f] = __expf(p[f] - mx); sum += p[f]; }
  const float inv = 1.0f / sum;

  unsigned short* op = Orecv + ((long)(g * CH + t)) * DMODEL + h * HD;
  for (int e = 0; e < HD; ++e) {
    float o = 0.0f;
#pragma unroll
    for (int f = 0; f < FACTOR; ++f) o += p[f] * sV[f][e];
    op[e] = f2bf(o * inv);
  }
}

// ---------------------------------------------------------------------------
// BatchNorm: y = a + b, column sums/sumsq (feature dim = 512), then apply.
// ---------------------------------------------------------------------------
__global__ void k_zero(float* __restrict__ p, int n)
{
  int i = blockIdx.x * blockDim.x + threadIdx.x;
  if (i < n) p[i] = 0.0f;
}

__global__ __launch_bounds__(256) void k_add_stats(
    const float* __restrict__ a, const float* __restrict__ b,
    float* __restrict__ y, float* __restrict__ sum, float* __restrict__ sq)
{
  const int t = threadIdx.x;
  const long base = (long)blockIdx.x * 128 * DMODEL;
  float s0 = 0, q0 = 0, s1 = 0, q1 = 0;
  for (int r = 0; r < 128; ++r) {
    long i0 = base + (long)r * DMODEL + t;
    long i1 = i0 + 256;
    float v0 = a[i0] + b[i0]; y[i0] = v0; s0 += v0; q0 += v0 * v0;
    float v1 = a[i1] + b[i1]; y[i1] = v1; s1 += v1; q1 += v1 * v1;
  }
  atomicAdd(&sum[t], s0);       atomicAdd(&sq[t], q0);
  atomicAdd(&sum[t + 256], s1); atomicAdd(&sq[t + 256], q1);
}

__global__ void k_bn_apply(const float* __restrict__ y,
                           const float* __restrict__ sum, const float* __restrict__ sq,
                           const float* __restrict__ gamma, const float* __restrict__ beta,
                           float* __restrict__ ef, unsigned short* __restrict__ eb)
{
  const float invCnt = 1.0f / (float)ROWS;
  const long total = (long)ROWS * DMODEL;
  for (long i = (long)blockIdx.x * blockDim.x + threadIdx.x; i < total;
       i += (long)gridDim.x * blockDim.x) {
    int d = (int)(i & (DMODEL - 1));
    float mean = sum[d] * invCnt;
    float var  = sq[d] * invCnt - mean * mean;
    float v = gamma[d] * (y[i] - mean) * rsqrtf(var + BN_EPS) + beta[d];
    ef[i] = v;
    eb[i] = f2bf(v);
  }
}

// ---------------------------------------------------------------------------
// host
// ---------------------------------------------------------------------------
extern "C" void kernel_launch(void* const* d_in, const int* in_sizes, int n_in,
                              void* d_out, int out_size, void* d_ws, size_t ws_size,
                              hipStream_t stream)
{
  const float* hiddens  = (const float*)d_in[0];
  const float* Wq       = (const float*)d_in[1];
  const float* bq       = (const float*)d_in[2];
  const float* Wk       = (const float*)d_in[3];
  const float* bk       = (const float*)d_in[4];
  const float* Wv       = (const float*)d_in[5];
  const float* bv       = (const float*)d_in[6];
  const float* Wo       = (const float*)d_in[7];
  const float* bo       = (const float*)d_in[8];
  const float* router   = (const float*)d_in[9];
  const float* bn_gamma = (const float*)d_in[10];
  const float* bn_beta  = (const float*)d_in[11];
  const float* mlp_W1   = (const float*)d_in[12];
  const float* mlp_b1   = (const float*)d_in[13];
  const float* mlp_W2   = (const float*)d_in[14];
  const float* mlp_b2   = (const float*)d_in[15];

  char* ws = (char*)d_ws;
  size_t off = 0;
  auto alloc = [&](size_t bytes) -> void* {
    off = (off + 255) & ~(size_t)255;
    void* p = ws + off;
    off += bytes;
    return p;
  };

  float*          dsF  = (float*)alloc((size_t)ROWS * DMODEL * 4);          // dim_send / e (f32)
  unsigned short* dsB  = (unsigned short*)alloc((size_t)ROWS * DMODEL * 2); // bf16 copy
  float*          R    = (float*)alloc((size_t)ROWS * DMODEL * 4);          // recv / mlp-out / BN y
  unsigned short* bufA = (unsigned short*)alloc((size_t)ROWS * DMODEL * 2); // Kd -> Qd -> h(chunk)
  unsigned short* bufB = (unsigned short*)alloc((size_t)ROWS * DMODEL * 2); // Vd -> arecv
  unsigned short* qr   = (unsigned short*)alloc((size_t)QRROWS * DMODEL * 2);
  unsigned short* Qr   = (unsigned short*)alloc((size_t)QRROWS * DMODEL * 2);
  unsigned short* abuf = (unsigned short*)alloc((size_t)BUFROWS * DMODEL * 2);
  unsigned short* bufT = (unsigned short*)alloc((size_t)BUFROWS * DMODEL * 2);
  unsigned short* Kb   = (unsigned short*)alloc((size_t)BUFROWS * DMODEL * 2);
  unsigned short* Vb   = (unsigned short*)alloc((size_t)BUFROWS * DMODEL * 2);
  unsigned short* Wt[8];
  for (int i = 0; i < 8; ++i)
    Wt[i] = (unsigned short*)alloc((size_t)DMODEL * DMODEL * 2);            // q1 k1 v1 o1 q2 k2 v2 o2
  unsigned short* W1t  = (unsigned short*)alloc((size_t)DMODEL * DFFN * 2);
  unsigned short* W2t  = (unsigned short*)alloc((size_t)DFFN * DMODEL * 2);
  float*          stS  = (float*)alloc(DMODEL * 4);
  float*          stQ  = (float*)alloc(DMODEL * 4);

  auto gemm = [&](const unsigned short* A, const unsigned short* Bt, const float* bias,
                  float* Cf, unsigned short* Cb, int M, int N, int K, int gelu) {
    dim3 g(N / 128, M / 128), b(128);
    k_gemm_bf16<<<g, b, 0, stream>>>(A, Bt, bias, Cf, Cb, M, N, K, gelu);
  };

  // input transpose + bf16 (x[b,c,n,d] -> dim layout rows (b,n,c))
  k_rearrange_in<<<8192, 256, 0, stream>>>(hiddens, dsF, dsB);

  for (int l = 0; l < NLAYERS; ++l) {
    const size_t dd = (size_t)DMODEL * DMODEL;
    // stage 1 = router->buf MHA, stage 2 = recv MHA (stage 0 is dead time MHA)
    const float* wsrc[8] = {
      Wq + ((size_t)l * 3 + 1) * dd, Wk + ((size_t)l * 3 + 1) * dd,
      Wv + ((size_t)l * 3 + 1) * dd, Wo + ((size_t)l * 3 + 1) * dd,
      Wq + ((size_t)l * 3 + 2) * dd, Wk + ((size_t)l * 3 + 2) * dd,
      Wv + ((size_t)l * 3 + 2) * dd, Wo + ((size_t)l * 3 + 2) * dd };
    for (int i = 0; i < 8; ++i)
      k_wtrans<<<1024, 256, 0, stream>>>(wsrc[i], Wt[i], DMODEL, DMODEL, (long)dd);
    k_wtrans<<<4096, 256, 0, stream>>>(mlp_W1 + ((size_t)l * 2 + 1) * DMODEL * DFFN,
                                       W1t, DMODEL, DFFN, (long)DMODEL * DFFN);
    k_wtrans<<<4096, 256, 0, stream>>>(mlp_W2 + ((size_t)l * 2 + 1) * DFFN * DMODEL,
                                       W2t, DFFN, DMODEL, (long)DFFN * DMODEL);
    k_cvt_bf16<<<1280, 256, 0, stream>>>(router + (size_t)l * QRROWS * DMODEL, qr,
                                         (long)QRROWS * DMODEL);

    const float* bq1 = bq + ((size_t)l * 3 + 1) * DMODEL;
    const float* bk1 = bk + ((size_t)l * 3 + 1) * DMODEL;
    const float* bv1 = bv + ((size_t)l * 3 + 1) * DMODEL;
    const float* bo1 = bo + ((size_t)l * 3 + 1) * DMODEL;
    const float* bq2 = bq + ((size_t)l * 3 + 2) * DMODEL;
    const float* bk2 = bk + ((size_t)l * 3 + 2) * DMODEL;
    const float* bv2 = bv + ((size_t)l * 3 + 2) * DMODEL;
    const float* bo2 = bo + ((size_t)l * 3 + 2) * DMODEL;

    // --- router->buf MHA ---
    gemm(qr,  Wt[0], bq1, nullptr, Qr,   QRROWS, DMODEL, DMODEL, 0);  // Q (router, dedup over b)
    gemm(dsB, Wt[1], bk1, nullptr, bufA, ROWS,   DMODEL, DMODEL, 0);  // K
    gemm(dsB, Wt[2], bv1, nullptr, bufB, ROWS,   DMODEL, DMODEL, 0);  // V
    k_attn_router<<<dim3(GROUPS, NHEADS), 64, 0, stream>>>(Qr, bufA, bufB, abuf);
    gemm(abuf, Wt[3], bo1, nullptr, bufT, BUFROWS, DMODEL, DMODEL, 0); // buf

    // --- recv MHA ---
    gemm(dsB,  Wt[4], bq2, nullptr, bufA, ROWS,    DMODEL, DMODEL, 0); // Q
    gemm(bufT, Wt[5], bk2, nullptr, Kb,   BUFROWS, DMODEL, DMODEL, 0); // K
    gemm(bufT, Wt[6], bv2, nullptr, Vb,   BUFROWS, DMODEL, DMODEL, 0); // V
    k_attn_recv<<<dim3(GROUPS, NHEADS), 64, 0, stream>>>(bufA, Kb, Vb, bufB);
    gemm(bufB, Wt[7], bo2, R, nullptr, ROWS, DMODEL, DMODEL, 0);       // recv (f32)

    // --- BN1: e = BN(dim_send + recv) ---
    const float* g2 = bn_gamma + ((size_t)l * 4 + 2) * DMODEL;
    const float* B2 = bn_beta  + ((size_t)l * 4 + 2) * DMODEL;
    k_zero<<<2, 256, 0, stream>>>(stS, DMODEL);
    k_zero<<<2, 256, 0, stream>>>(stQ, DMODEL);
    k_add_stats<<<ROWS / 128, 256, 0, stream>>>(dsF, R, R, stS, stQ);
    k_bn_apply<<<8192, 256, 0, stream>>>(R, stS, stQ, g2, B2, dsF, dsB);

    // --- MLP (chunked): m = GELU(e @ W1 + b1) @ W2 + b2 ---
    const float* b1p = mlp_b1 + ((size_t)l * 2 + 1) * DFFN;
    const float* b2p = mlp_b2 + ((size_t)l * 2 + 1) * DMODEL;
    for (int c = 0; c < ROWS / MLP_CHUNK; ++c) {
      gemm(dsB + (size_t)c * MLP_CHUNK * DMODEL, W1t, b1p,
           nullptr, bufA, MLP_CHUNK, DFFN, DMODEL, 1);
      gemm(bufA, W2t, b2p,
           R + (size_t)c * MLP_CHUNK * DMODEL, nullptr, MLP_CHUNK, DMODEL, DFFN, 0);
    }

    // --- BN2: out = BN(e + m); becomes next layer's dim_send directly ---
    const float* g3 = bn_gamma + ((size_t)l * 4 + 3) * DMODEL;
    const float* B3 = bn_beta  + ((size_t)l * 4 + 3) * DMODEL;
    k_zero<<<2, 256, 0, stream>>>(stS, DMODEL);
    k_zero<<<2, 256, 0, stream>>>(stQ, DMODEL);
    k_add_stats<<<ROWS / 128, 256, 0, stream>>>(dsF, R, R, stS, stQ);
    k_bn_apply<<<8192, 256, 0, stream>>>(R, stS, stQ, g3, B3, dsF, dsB);
  }

  k_rearrange_out<<<8192, 256, 0, stream>>>(dsF, (float*)d_out);
}
